// ETCContext_3332894621935
// MI455X (gfx1250) — matrix operand order repaired and checked by reading, not compiled, not run on hardware
//
#include <hip/hip_runtime.h>
#include <hip/hip_bf16.h>

// Problem constants (from reference)
#define BB 4
#define LL 1024
#define DD 512
#define HH 8
#define DH 64          // head dim
#define EPSN 1e-8f
#define LN_EPS 1e-5f

typedef __attribute__((ext_vector_type(16))) _Float16 v16h;
typedef __attribute__((ext_vector_type(8)))  _Float16 v8h;
typedef __attribute__((ext_vector_type(8)))  float    v8f;

// ---------------------------------------------------------------------------
// Kernel 1: split heads, L2-normalize K and Q, convert to f16.
//   Kn  [bh][L][64]  (row-major, WMMA A operand for scores GEMM)
//   QnT [bh][64][L]  (transposed: WMMA B operand rows are K-contiguous)
//   Vh  [bh][L][64]  (row-major = K x N for the output GEMM B operand)
// One wave per (b,h,l) row: 64 elems -> 2 per lane, shuffle reduction.
// ---------------------------------------------------------------------------
__global__ void etc_prep_kernel(const float* __restrict__ K,
                                const float* __restrict__ Q,
                                const float* __restrict__ V,
                                _Float16* __restrict__ Kn,
                                _Float16* __restrict__ QnT,
                                _Float16* __restrict__ Vh) {
  const int lane = threadIdx.x;                       // 0..31
  const long row = (long)blockIdx.x * blockDim.y + threadIdx.y; // 0..B*H*L-1
  const int l  = (int)(row & (LL - 1));
  const int bh = (int)(row >> 10);                    // b*H + h
  const int h  = bh & (HH - 1);
  const int b  = bh >> 3;

  const float* kp = K + ((long)b * LL + l) * DD + h * DH;
  const float* qp = Q + ((long)b * LL + l) * DD + h * DH;
  const float* vp = V + ((long)b * LL + l) * DD + h * DH;

  const int d0 = 2 * lane;
  float k0 = kp[d0], k1 = kp[d0 + 1];
  float q0 = qp[d0], q1 = qp[d0 + 1];
  float v0 = vp[d0], v1 = vp[d0 + 1];

  float ks = k0 * k0 + k1 * k1;
  float qs = q0 * q0 + q1 * q1;
#pragma unroll
  for (int m = 16; m >= 1; m >>= 1) {
    ks += __shfl_xor(ks, m, 32);
    qs += __shfl_xor(qs, m, 32);
  }
  const float kr = 1.0f / fmaxf(sqrtf(ks), EPSN);  // clamp(norm, EPS)
  const float qr = 1.0f / fmaxf(sqrtf(qs), EPSN);

  _Float16* kd = Kn + ((long)bh * LL + l) * DH;
  kd[d0]     = (_Float16)(k0 * kr);
  kd[d0 + 1] = (_Float16)(k1 * kr);

  _Float16* qd = QnT + (long)bh * DH * LL;           // [64][L]
  qd[(long)d0 * LL + l]       = (_Float16)(q0 * qr);
  qd[(long)(d0 + 1) * LL + l] = (_Float16)(q1 * qr);

  _Float16* vd = Vh + ((long)bh * LL + l) * DH;
  vd[d0]     = (_Float16)v0;
  vd[d0 + 1] = (_Float16)v1;
}

// ---------------------------------------------------------------------------
// WMMA fragment loaders (layouts per CDNA5 ISA 7.12.2, wave32).
// A (16x32 f16, M x K): lane<16 -> row=lane, K {0..7}U{16..23};
//                       lane>=16 -> row=lane-16, K {8..15}U{24..31}.
// B (32x16 f16, K x N): lane l -> col N=l&15, K same as A for its lane half.
// ---------------------------------------------------------------------------
__device__ __forceinline__ v16h load_fragA(const _Float16* base, long row_stride,
                                           int kk, int lane) {
  const int arow = lane & 15;
  const int koff = (lane >> 4) * 8;
  v16h a;
  *(v8h*)&a       = *(const v8h*)(base + (long)arow * row_stride + kk + koff);
  *((v8h*)&a + 1) = *(const v8h*)(base + (long)arow * row_stride + kk + koff + 16);
  return a;
}

__device__ __forceinline__ v16h load_fragB(const _Float16* base, long row_stride,
                                           int kk, int n0, int lane) {
  const int bcol = lane & 15;
  const int koff = (lane >> 4) * 8;
  v16h b;
#pragma unroll
  for (int s = 0; s < 8; ++s) {
    b[s]     = base[(long)(kk + koff + s) * row_stride + n0 + bcol];
    b[s + 8] = base[(long)(kk + koff + 16 + s) * row_stride + n0 + bcol];
  }
  return b;
}

// ---------------------------------------------------------------------------
// Kernel 2: scores GEMM (co = Kn . Qn^T), head-axis LayerNorm in registers,
// pad mask -> -inf, store f16 scores S [bh][L][L].
// One wave per 16x16 (i,j) tile; 8 heads x 2 k-steps = 16 WMMAs per wave.
// ---------------------------------------------------------------------------
__global__ void etc_scores_kernel(const _Float16* __restrict__ Kn,
                                  const _Float16* __restrict__ QnT,
                                  const float* __restrict__ ln_w,
                                  const float* __restrict__ ln_b,
                                  const unsigned char* __restrict__ pad_mask,
                                  _Float16* __restrict__ S) {
  const int lane = threadIdx.x;
  const int b  = blockIdx.z;
  const int i0 = blockIdx.y * 16;
  const int j0 = (blockIdx.x * 4 + threadIdx.y) * 16;

  float acc[HH][8];

#pragma unroll
  for (int h = 0; h < HH; ++h) {
    const int bh = b * HH + h;
    const _Float16* A  = Kn  + ((long)bh * LL + i0) * DH;
    const _Float16* Bp = QnT + (long)bh * DH * LL;
    v8f c = {};
#pragma unroll
    for (int kk = 0; kk < DH; kk += 32) {
      v16h a  = load_fragA(A, DH, kk, lane);
      v16h bf = load_fragB(Bp, LL, kk, j0, lane);
      c = __builtin_amdgcn_wmma_f32_16x16x32_f16(false, a, false, bf,
                                                 (short)0, c, false, false);
    }
#pragma unroll
    for (int r = 0; r < 8; ++r) acc[h][r] = c[r];
  }

  float lw[HH], lb[HH];
#pragma unroll
  for (int h = 0; h < HH; ++h) { lw[h] = ln_w[h]; lb[h] = ln_b[h]; }

  const int m_off = (lane >> 4) * 8;
  const int n     = lane & 15;

#pragma unroll
  for (int r = 0; r < 8; ++r) {
    // LayerNorm across the 8 heads: pointwise in (r, lane).
    float mu = 0.0f;
#pragma unroll
    for (int h = 0; h < HH; ++h) mu += acc[h][r];
    mu *= 0.125f;
    float var = 0.0f;
#pragma unroll
    for (int h = 0; h < HH; ++h) { float t = acc[h][r] - mu; var += t * t; }
    var *= 0.125f;
    const float inv = rsqrtf(var + LN_EPS);

    const int m = r + m_off;
    const bool keep =
        pad_mask[((long)b * LL + (i0 + m)) * LL + (j0 + n)] != 0;

#pragma unroll
    for (int h = 0; h < HH; ++h) {
      float y = (acc[h][r] - mu) * inv * lw[h] + lb[h];
      if (!keep) y = -__builtin_inff();
      S[((long)(b * HH + h) * LL + (i0 + m)) * LL + (j0 + n)] = (_Float16)y;
    }
  }
}

// ---------------------------------------------------------------------------
// Kernel 3: in-place softmax over keys for each (b,h,i) row of S (f16).
// All-(-inf) rows -> 0 (matches reference's NaN cleanup).
// ---------------------------------------------------------------------------
__global__ void etc_softmax_kernel(_Float16* __restrict__ S) {
  __shared__ float red[256];
  const long row = blockIdx.x;                 // B*H*L rows
  _Float16* p = S + row * LL;
  const int t = threadIdx.x;                   // 256 threads, 4 elems each

  float x[4];
#pragma unroll
  for (int u = 0; u < 4; ++u) x[u] = (float)p[t + 256 * u];

  float mx = fmaxf(fmaxf(x[0], x[1]), fmaxf(x[2], x[3]));
  red[t] = mx; __syncthreads();
  for (int s = 128; s > 0; s >>= 1) {
    if (t < s) red[t] = fmaxf(red[t], red[t + s]);
    __syncthreads();
  }
  const float m = red[0]; __syncthreads();

  float s = 0.0f;
  const bool dead = (m == -__builtin_inff());
#pragma unroll
  for (int u = 0; u < 4; ++u) {
    float e = dead ? 0.0f : __expf(x[u] - m);
    x[u] = e; s += e;
  }
  red[t] = s; __syncthreads();
  for (int q = 128; q > 0; q >>= 1) {
    if (t < q) red[t] += red[t + q];
    __syncthreads();
  }
  const float tot = red[0];
  const float r = (tot > 0.0f) ? (1.0f / tot) : 0.0f;
#pragma unroll
  for (int u = 0; u < 4; ++u) p[t + 256 * u] = (_Float16)(x[u] * r);
}

// ---------------------------------------------------------------------------
// Async global->LDS helpers (CDNA5 GLOBAL_LOAD_ASYNC_TO_LDS_*, ASYNCcnt).
// Inline asm per CDNA5 ISA (portable across toolchains).
// VDST VGPR carries the LDS byte address; vaddr is the 64-bit global address.
// ---------------------------------------------------------------------------
__device__ __forceinline__ void async_load_b64_to_lds(const void* gsrc,
                                                      void* lds_dst) {
  unsigned lds_off = (unsigned)(size_t)lds_dst;            // LDS aperture low bits
  unsigned long long ga = (unsigned long long)(size_t)gsrc;
  asm volatile("global_load_async_to_lds_b64 %0, %1, off"
               :
               : "v"(lds_off), "v"(ga)
               : "memory");
}

__device__ __forceinline__ void wait_asynccnt0() {
  asm volatile("s_wait_asynccnt 0x0" ::: "memory");
}

// ---------------------------------------------------------------------------
// Kernel 4a: partial column sums of P with double-buffered async row staging.
// Block = (b, h, i-chunk of 128 rows), 256 threads. Each iteration: the next
// row is streamed Global->LDS asynchronously (B64 per thread) while the
// current LDS row is reduced into per-thread partials; one f32 atomic per
// (thread, column) at the end. colsum must be pre-zeroed.
// ---------------------------------------------------------------------------
#define ICHUNK 128
__global__ void etc_colsum_kernel(const _Float16* __restrict__ S,
                                  float* __restrict__ colsum) {
  __shared__ _Float16 sbuf[2][LL];             // 2 x 2KB row buffers
  const int t   = threadIdx.x;                 // 0..255
  const int blk = blockIdx.x;                  // B*H*(L/ICHUNK) = 256
  const int chunk = blk & (LL / ICHUNK - 1);   // 0..7
  const int bh    = blk >> 3;                  // 0..31
  const int b     = bh >> 3;
  const int i0    = chunk * ICHUNK;

  const _Float16* base = S + ((long)bh * LL + i0) * LL;

  // prologue: stage row 0 of the chunk into buffer 0 (8 bytes per thread)
  async_load_b64_to_lds(base + t * 4, &sbuf[0][t * 4]);

  float part[4] = {0.f, 0.f, 0.f, 0.f};
  for (int i = 0; i < ICHUNK; ++i) {
    wait_asynccnt0();
    __syncthreads();                           // staged row visible to all waves
    if (i + 1 < ICHUNK)                        // kick next row into other buffer
      async_load_b64_to_lds(base + (long)(i + 1) * LL + t * 4,
                            &sbuf[(i + 1) & 1][t * 4]);
    const _Float16* rowp = sbuf[i & 1];
#pragma unroll
    for (int u = 0; u < 4; ++u) part[u] += (float)rowp[t + 256 * u];
  }

  float* cs = colsum + b * LL;
#pragma unroll
  for (int u = 0; u < 4; ++u) atomicAdd(cs + t + 256 * u, part[u]);
}

__global__ void etc_zero_kernel(float* __restrict__ p) {
  p[blockIdx.x * blockDim.x + threadIdx.x] = 0.0f;
}

// ---------------------------------------------------------------------------
// Kernel 4b: w = softmax_j(mask(colsum/doc_size)), written to output tail.
// One block per batch.
// ---------------------------------------------------------------------------
__global__ void etc_wsoftmax_kernel(const float* __restrict__ colsum,
                                    const float* __restrict__ doc_sizes,
                                    const unsigned char* __restrict__ bx_packed,
                                    float* __restrict__ wout) {
  __shared__ float red[256];
  const int b = blockIdx.x;
  const int t = threadIdx.x;
  const float inv_doc = 1.0f / doc_sizes[b];

  float x[4];
#pragma unroll
  for (int u = 0; u < 4; ++u) {
    const int j = t + 256 * u;
    float v = colsum[b * LL + j] * inv_doc;
    if (bx_packed[b * LL + j]) v = -__builtin_inff();
    x[u] = v;
  }
  float mx = fmaxf(fmaxf(x[0], x[1]), fmaxf(x[2], x[3]));
  red[t] = mx; __syncthreads();
  for (int s = 128; s > 0; s >>= 1) {
    if (t < s) red[t] = fmaxf(red[t], red[t + s]);
    __syncthreads();
  }
  const float m = red[0]; __syncthreads();

  float s = 0.0f;
  const bool dead = (m == -__builtin_inff());
#pragma unroll
  for (int u = 0; u < 4; ++u) {
    float e = dead ? 0.0f : __expf(x[u] - m);
    x[u] = e; s += e;
  }
  red[t] = s; __syncthreads();
  for (int q = 128; q > 0; q >>= 1) {
    if (t < q) red[t] += red[t + q];
    __syncthreads();
  }
  const float tot = red[0];
  const float r = (tot > 0.0f) ? (1.0f / tot) : 0.0f;
#pragma unroll
  for (int u = 0; u < 4; ++u) wout[b * LL + t + 256 * u] = x[u] * r;
}

// ---------------------------------------------------------------------------
// Kernel 5: out = P . Vh  (per (b,h): [L,L]x[L,64]), merge heads on store.
// One wave per (i-tile, d-tile); 32 WMMA k-steps over j.
// ---------------------------------------------------------------------------
__global__ void etc_outgemm_kernel(const _Float16* __restrict__ S,
                                   const _Float16* __restrict__ Vh,
                                   float* __restrict__ out) {
  const int lane = threadIdx.x;
  const int d0 = threadIdx.y * 16;
  const int i0 = blockIdx.x * 16;
  const int h  = blockIdx.y;
  const int b  = blockIdx.z;
  const int bh = b * HH + h;

  const _Float16* A  = S  + ((long)bh * LL + i0) * LL;
  const _Float16* Bp = Vh + (long)bh * LL * DH;

  v8f c = {};
  for (int kk = 0; kk < LL; kk += 32) {
    v16h a  = load_fragA(A, LL, kk, lane);
    v16h bf = load_fragB(Bp, DH, kk, d0, lane);
    if (kk + 64 < LL)   // stream the V panel through L2 ahead of use
      __builtin_prefetch((const void*)(Bp + (long)(kk + 64 + lane) * DH + d0), 0, 0);
    c = __builtin_amdgcn_wmma_f32_16x16x32_f16(false, a, false, bf,
                                               (short)0, c, false, false);
  }

  const int m_off = (lane >> 4) * 8;
  const int n     = lane & 15;
#pragma unroll
  for (int r = 0; r < 8; ++r) {
    const int m = r + m_off;
    out[((long)b * LL + (i0 + m)) * DD + h * DH + d0 + n] = c[r];
  }
}

// ---------------------------------------------------------------------------
// Host launcher. Inputs (setup_inputs order):
//   0:K f32  1:Q f32  2:V f32  3:doc_sizes f32[B]  4:ln_w f32[H]  5:ln_b f32[H]
//   6:pad_mask bool[B,L,L]  7:bx_packed bool[B,L]
// Output: out f32[B,L,D] ++ w f32[B,L] (flattened tuple).
// Workspace layout: Kn 4MB | QnT 4MB | Vh 4MB | S 64MB | colsum 16KB (~76MB)
// ---------------------------------------------------------------------------
extern "C" void kernel_launch(void* const* d_in, const int* in_sizes, int n_in,
                              void* d_out, int out_size, void* d_ws, size_t ws_size,
                              hipStream_t stream) {
  const float* K  = (const float*)d_in[0];
  const float* Q  = (const float*)d_in[1];
  const float* V  = (const float*)d_in[2];
  const float* doc = (const float*)d_in[3];
  const float* lnw = (const float*)d_in[4];
  const float* lnb = (const float*)d_in[5];
  const unsigned char* pad = (const unsigned char*)d_in[6];
  const unsigned char* bx  = (const unsigned char*)d_in[7];
  float* out = (float*)d_out;

  char* ws = (char*)d_ws;
  _Float16* Kn  = (_Float16*)(ws);
  _Float16* QnT = (_Float16*)(ws + ((size_t)4 << 20));
  _Float16* Vh  = (_Float16*)(ws + ((size_t)8 << 20));
  _Float16* S   = (_Float16*)(ws + ((size_t)12 << 20));
  float* colsum = (float*)(ws + ((size_t)76 << 20));

  float* wout = out + (size_t)BB * LL * DD;

  // 1: prep (B*H*L rows, 4 waves/block)
  etc_prep_kernel<<<dim3((BB * HH * LL) / 4), dim3(32, 4), 0, stream>>>(
      K, Q, V, Kn, QnT, Vh);

  // 2: scores GEMM + head-LN + mask (16x16 tile per wave)
  etc_scores_kernel<<<dim3(LL / 64, LL / 16, BB), dim3(32, 4), 0, stream>>>(
      Kn, QnT, lnw, lnb, pad, S);

  // 3: row softmax in place
  etc_softmax_kernel<<<dim3(BB * HH * LL), dim3(256), 0, stream>>>(S);

  // 4a: zero accumulator, then partial column sums (async LDS staging)
  etc_zero_kernel<<<dim3((BB * LL) / 256), dim3(256), 0, stream>>>(colsum);
  etc_colsum_kernel<<<dim3(BB * HH * (LL / ICHUNK)), dim3(256), 0, stream>>>(
      S, colsum);

  // 4b: doc-level softmax -> output tail
  etc_wsoftmax_kernel<<<dim3(BB), dim3(256), 0, stream>>>(colsum, doc, bx, wout);

  // 5: output GEMM, merge heads
  etc_outgemm_kernel<<<dim3(LL / 16, HH, BB), dim3(32, 4), 0, stream>>>(
      S, Vh, out);
}
